// Attention_75539884802796
// MI455X (gfx1250) — matrix operand order repaired and checked by reading, not compiled, not run on hardware
//
#include <hip/hip_runtime.h>
#include <hip/hip_bf16.h>

// ---------------------------------------------------------------------------
// GQA attention for MI455X (gfx1250, wave32, WMMA).
//   out = softmax(mask(RoPE(xWq) RoPE(xWk)^T * d^-1/2)) (xWv)  @ Wo
// All matmuls via v_wmma_f32_16x16x32_f16 (f32 accumulate); flash-attention
// style fused softmax; GEMM tiles staged with async global->LDS copies
// (ASYNCcnt) and double buffering when the toolchain exposes the builtin.
// ---------------------------------------------------------------------------

typedef __attribute__((ext_vector_type(8)))  _Float16 v8h;
typedef __attribute__((ext_vector_type(16))) _Float16 v16h;
typedef __attribute__((ext_vector_type(8)))  float    v8f;
typedef int v4i_vs __attribute__((vector_size(16)));   // matches builtin proto

#define B_SZ     8
#define S_LEN    512
#define HID      2048
#define NH       32
#define NKV      8
#define HD       64
#define GRP      (NH / NKV)     // 4

#define AS1 __attribute__((address_space(1)))
#define AS3 __attribute__((address_space(3)))

#if defined(__has_builtin)
#  if __has_builtin(__builtin_amdgcn_global_load_async_to_lds_b128)
#    define HAVE_ASYNC_COPY 1
#  else
#    define HAVE_ASYNC_COPY 0
#  endif
#else
#  define HAVE_ASYNC_COPY 0
#endif

// 16-byte global -> LDS copy: async (no VGPR bounce, tracked by ASYNCcnt)
// when available, else plain load+store (compiler inserts dscnt waits).
__device__ __forceinline__ void copy16_g2l(const _Float16* g, _Float16* lds) {
#if HAVE_ASYNC_COPY
  __builtin_amdgcn_global_load_async_to_lds_b128((AS1 v4i_vs*)g,
                                                 (AS3 v4i_vs*)lds, 0, 0);
#else
  *reinterpret_cast<v8h*>(lds) = *reinterpret_cast<const v8h*>(g);
#endif
}

template <int N>
__device__ __forceinline__ void wait_async() {
#if HAVE_ASYNC_COPY
#  if __has_builtin(__builtin_amdgcn_s_wait_asynccnt)
  __builtin_amdgcn_s_wait_asynccnt(N);
#  else
  asm volatile("s_wait_asynccnt %0" :: "n"(N) : "memory");
#  endif
#endif
}

__device__ __forceinline__ v8f v8f_zero() {
  v8f z;
#pragma unroll
  for (int i = 0; i < 8; ++i) z[i] = 0.0f;
  return z;
}

// A-fragment (16x32 f16, M x K) from a row-major tile, stride ld (halves).
// Lane L (0..15): row M=L, K = {0..7, 16..23}; lane L+16: row M=L, K = {8..15, 24..31}.
__device__ __forceinline__ v16h load_frag_a(const _Float16* tile, int ld) {
  const int lane = threadIdx.x & 31;
  const int m    = lane & 15;
  const int kb   = (lane >> 4) << 3;            // 0 or 8
  const _Float16* p = tile + m * ld + kb;
  v8h lo = *reinterpret_cast<const v8h*>(p);      // K = kb .. kb+7
  v8h hi = *reinterpret_cast<const v8h*>(p + 16); // K = kb+16 .. kb+23
  return __builtin_shufflevector(lo, hi, 0, 1, 2, 3, 4, 5, 6, 7,
                                 8, 9, 10, 11, 12, 13, 14, 15);
}

// B-fragment (32x16 f16, K x N) from a row-major tile, stride ld (halves).
// Lane L (0..15): column N=L, K = {0..7, 16..23}; lane L+16: column N=L, K = {8..15, 24..31}.
__device__ __forceinline__ v16h load_frag_b(const _Float16* tile, int ld) {
  const int lane = threadIdx.x & 31;
  const int n    = lane & 15;
  const int kb   = (lane >> 4) << 3;
  const _Float16* p = tile + kb * ld + n;
  v16h r;
#pragma unroll
  for (int i = 0; i < 8; ++i) {
    r[i]     = p[i * ld];
    r[8 + i] = p[(16 + i) * ld];
  }
  return r;
}

#define WMMA_F16(A, Bm, C) \
  __builtin_amdgcn_wmma_f32_16x16x32_f16(false, (A), false, (Bm), (short)0, (C), false, false)

// ---------------------------------------------------------------------------
// f32 -> f16 cast (grid-stride)
// ---------------------------------------------------------------------------
__global__ void __launch_bounds__(256) cvt_f16_kernel(const float* __restrict__ in,
                                                      _Float16* __restrict__ out, long n) {
  long i = (long)blockIdx.x * blockDim.x + threadIdx.x;
  const long stride = (long)gridDim.x * blockDim.x;
  for (; i < n; i += stride) out[i] = (_Float16)in[i];
}

// ---------------------------------------------------------------------------
// GEMM: C[M,N] (f32) = A[M,K] (f16, row-major) * B[K,N] (f16, row-major)
// 128x128 block tile, K-step 32, 256 threads = 8 waves in a 4x2 wave grid,
// each wave owns a 32x64 sub-tile = 2x4 WMMA accumulators.
// Double-buffered LDS; tiles staged with async global->LDS copies.
// M,N,K assumed multiples of 128/128/32 (true for all calls here).
// ---------------------------------------------------------------------------
__global__ void __launch_bounds__(256) gemm_f16_f32(const _Float16* __restrict__ A,
                                                    const _Float16* __restrict__ Bm,
                                                    float* __restrict__ C,
                                                    int M, int N, int K) {
  __shared__ __align__(16) _Float16 As[2][128 * 32];
  __shared__ __align__(16) _Float16 Bs[2][32 * 128];

  const int t    = threadIdx.x;
  const int wave = t >> 5;
  const int lane = t & 31;
  const int wm   = wave >> 1;          // 0..3
  const int wn   = wave & 1;           // 0..1
  const int row0 = blockIdx.y * 128;
  const int col0 = blockIdx.x * 128;

  v8f acc[2][4];
#pragma unroll
  for (int i = 0; i < 2; ++i)
#pragma unroll
    for (int j = 0; j < 4; ++j) acc[i][j] = v8f_zero();

  // Per-thread staging slots: 2x b128 for A (128x32 tile), 2x b128 for B (32x128).
  const int arow = t >> 1, acol = (t & 1) * 16;
  const int brow = t >> 3, bcol = (t & 7) * 16;
  const _Float16* gA = A + (long)(row0 + arow) * K + acol;   // + kt*32
  const _Float16* gB = Bm + (long)brow * N + col0 + bcol;    // + kt*32*N
  const int la = arow * 32 + acol;
  const int lb = brow * 128 + bcol;

  const int KT = K / 32;

  // stage 0 prefetch
  copy16_g2l(gA,     &As[0][la]);
  copy16_g2l(gA + 8, &As[0][la + 8]);
  copy16_g2l(gB,     &Bs[0][lb]);
  copy16_g2l(gB + 8, &Bs[0][lb + 8]);

  for (int kt = 0; kt < KT; ++kt) {
    const int cur = kt & 1;
    if (kt + 1 < KT) {
      const _Float16* nA = gA + (kt + 1) * 32;
      const _Float16* nB = gB + (long)(kt + 1) * 32 * N;
      copy16_g2l(nA,     &As[cur ^ 1][la]);
      copy16_g2l(nA + 8, &As[cur ^ 1][la + 8]);
      copy16_g2l(nB,     &Bs[cur ^ 1][lb]);
      copy16_g2l(nB + 8, &Bs[cur ^ 1][lb + 8]);
      wait_async<4>();   // current stage's 4 older async ops are done
    } else {
      wait_async<0>();
    }
    __syncthreads();

    v16h a0 = load_frag_a(&As[cur][(wm * 32) * 32], 32);
    v16h a1 = load_frag_a(&As[cur][(wm * 32 + 16) * 32], 32);
#pragma unroll
    for (int j = 0; j < 4; ++j) {
      v16h b = load_frag_b(&Bs[cur][wn * 64 + j * 16], 128);
      acc[0][j] = WMMA_F16(a0, b, acc[0][j]);
      acc[1][j] = WMMA_F16(a1, b, acc[1][j]);
    }
    __syncthreads();     // buf `cur` gets overwritten in the next iteration
  }

  const int cn = lane & 15;
  const int mg = (lane >> 4) * 8;
#pragma unroll
  for (int i = 0; i < 2; ++i)
#pragma unroll
    for (int j = 0; j < 4; ++j)
#pragma unroll
      for (int r = 0; r < 8; ++r) {
        const int row = row0 + wm * 32 + i * 16 + mg + r;
        const int col = col0 + wn * 64 + j * 16 + cn;
        C[(long)row * N + col] = acc[i][j][r];
      }
}

// ---------------------------------------------------------------------------
// RoPE helpers.  inv_freq[i] = 10000^(-i/32) = exp(-ln(10000)/32 * i)
// rotate_half: rot[d] = d<32 ? -x[d+32] : x[d-32]; angle uses i = d & 31.
// ---------------------------------------------------------------------------
#define LN1E4_OVER_32 0.28782313662425572f

// Q: [B,S,H*D] f32 -> roped f16 [B,H,S,D]
__global__ void __launch_bounds__(256) rope_q_kernel(const float* __restrict__ q,
                                                     _Float16* __restrict__ qh) {
  const long n = (long)B_SZ * S_LEN * NH * HD;
  long e = (long)blockIdx.x * blockDim.x + threadIdx.x;
  if (e >= n) return;
  const int d = (int)(e & 63);
  const int h = (int)((e >> 6) & 31);
  const int s = (int)((e >> 11) & 511);
  const int b = (int)(e >> 20);
  const float x  = q[e];
  const float xr = (d < 32) ? -q[e + 32] : q[e - 32];
  const float ang = (float)s * __expf(-LN1E4_OVER_32 * (float)(d & 31));
  float sn, cs;
  __sincosf(ang, &sn, &cs);
  qh[(((long)(b * NH + h) * S_LEN) + s) * HD + d] = (_Float16)(x * cs + xr * sn);
}

// K: [B,S,Hk*D] f32 -> roped f32 out [B,Hk,S,D] and roped f16 K^T [B,Hk,D,S]
__global__ void __launch_bounds__(256) rope_k_kernel(const float* __restrict__ k,
                                                     float* __restrict__ kout,
                                                     _Float16* __restrict__ kt) {
  const long n = (long)B_SZ * S_LEN * NKV * HD;
  long e = (long)blockIdx.x * blockDim.x + threadIdx.x;
  if (e >= n) return;
  const int d = (int)(e & 63);
  const int h = (int)((e >> 6) & 7);
  const int s = (int)((e >> 9) & 511);
  const int b = (int)(e >> 18);
  const float x  = k[e];
  const float xr = (d < 32) ? -k[e + 32] : k[e - 32];
  const float ang = (float)s * __expf(-LN1E4_OVER_32 * (float)(d & 31));
  float sn, cs;
  __sincosf(ang, &sn, &cs);
  const float r = x * cs + xr * sn;
  kout[(((long)(b * NKV + h) * S_LEN) + s) * HD + d] = r;
  kt[(((long)(b * NKV + h) * HD) + d) * S_LEN + s]   = (_Float16)r;
}

// V: [B,S,Hk*D] f32 -> f32 out [B,Hk,S,D] and f16 [B,Hk,S,D]
__global__ void __launch_bounds__(256) copy_v_kernel(const float* __restrict__ v,
                                                     float* __restrict__ vout,
                                                     _Float16* __restrict__ vh) {
  const long n = (long)B_SZ * S_LEN * NKV * HD;
  long e = (long)blockIdx.x * blockDim.x + threadIdx.x;
  if (e >= n) return;
  const int d = (int)(e & 63);
  const int h = (int)((e >> 6) & 7);
  const int s = (int)((e >> 9) & 511);
  const int b = (int)(e >> 18);
  const long o = (((long)(b * NKV + h) * S_LEN) + s) * HD + d;
  const float x = v[e];
  vout[o] = x;
  vh[o]   = (_Float16)x;
}

// ---------------------------------------------------------------------------
// Fused flash attention.
// Grid: (8 query tiles of 64, B*NH).  Block: 128 threads = 4 waves.
// Wave w owns 16 query rows; online softmax; causal mask; WMMA for QK^T & PV.
// Q: [B,NH,S,D] f16;  Kt: [B,NKV,D,S] f16;  V: [B,NKV,S,D] f16.
// Output (pre-Wo): [B,S,NH*D] f16.
// ---------------------------------------------------------------------------
__global__ void __launch_bounds__(128) attn_kernel(const _Float16* __restrict__ Qh,
                                                   const _Float16* __restrict__ Kt,
                                                   const _Float16* __restrict__ Vh,
                                                   _Float16* __restrict__ Oh) {
  __shared__ __align__(16) _Float16 Ps[4][16 * 64];   // per-wave P tile

  const int t    = threadIdx.x;
  const int w    = t >> 5;
  const int lane = t & 31;
  const int qt   = blockIdx.x;              // query tile 0..7
  const int bh   = blockIdx.y;              // 0..255
  const int b    = bh >> 5;
  const int h    = bh & 31;
  const int hk   = h / GRP;
  const int qrow0 = qt * 64 + w * 16;

  const int cn = lane & 15;
  const int mg = (lane >> 4) * 8;

  const _Float16* qptr   = Qh + (((long)(b * NH + h) * S_LEN) + qrow0) * HD;
  const _Float16* ktbase = Kt + ((long)(b * NKV + hk) * HD) * S_LEN;
  const _Float16* vbase  = Vh + ((long)(b * NKV + hk) * S_LEN) * HD;

  const v16h aq0 = load_frag_a(qptr, HD);        // K (=d) 0..31
  const v16h aq1 = load_frag_a(qptr + 32, HD);   // K (=d) 32..63

  float m_i[8], l_i[8];
  v8f o[4];
#pragma unroll
  for (int r = 0; r < 8; ++r) { m_i[r] = -1e30f; l_i[r] = 0.0f; }
#pragma unroll
  for (int n = 0; n < 4; ++n) o[n] = v8f_zero();

  const int nkv = qt + 1;                        // causal: kv blocks 0..qt
  for (int jb = 0; jb < nkv; ++jb) {
    const int j0 = jb * 64;

    // prefetch next KV block while this one computes (global_prefetch_b8)
    if (jb + 1 < nkv) {
      const int jn0 = j0 + 64;
      __builtin_prefetch(ktbase + (long)lane * S_LEN + jn0, 0, 0);
      __builtin_prefetch(ktbase + (long)(32 + lane) * S_LEN + jn0, 0, 0);
      __builtin_prefetch(vbase + (long)(jn0 + lane) * HD, 0, 0);
      __builtin_prefetch(vbase + (long)(jn0 + 32 + lane) * HD, 0, 0);
    }

    // ---- S = (Q K^T) * scale, masked -------------------------------------
    v8f s[4];
#pragma unroll
    for (int n = 0; n < 4; ++n) {
      v16h bk0 = load_frag_b(ktbase + j0 + n * 16, S_LEN);              // d 0..31
      v16h bk1 = load_frag_b(ktbase + 32 * S_LEN + j0 + n * 16, S_LEN); // d 32..63
      v8f z = v8f_zero();
      z = WMMA_F16(aq0, bk0, z);
      z = WMMA_F16(aq1, bk1, z);
      s[n] = z;
    }

    float mnew[8];
#pragma unroll
    for (int r = 0; r < 8; ++r) mnew[r] = m_i[r];
#pragma unroll
    for (int n = 0; n < 4; ++n) {
      const int col = j0 + n * 16 + cn;
#pragma unroll
      for (int r = 0; r < 8; ++r) {
        const int qi = qrow0 + mg + r;
        float vv = s[n][r] * 0.125f;               // HD^-0.5
        vv = (col > qi) ? -1e30f : vv;
        s[n][r] = vv;
        mnew[r] = fmaxf(mnew[r], vv);
      }
    }
    // row-max across the 16 lanes holding one row
#pragma unroll
    for (int off = 1; off < 16; off <<= 1)
#pragma unroll
      for (int r = 0; r < 8; ++r)
        mnew[r] = fmaxf(mnew[r], __shfl_xor(mnew[r], off, 32));

    float rescale[8], lsum[8];
#pragma unroll
    for (int r = 0; r < 8; ++r) {
      rescale[r] = __expf(m_i[r] - mnew[r]);
      m_i[r] = mnew[r];
      lsum[r] = 0.0f;
    }
#pragma unroll
    for (int n = 0; n < 4; ++n)
#pragma unroll
      for (int r = 0; r < 8; ++r) {
        const float p = __expf(s[n][r] - mnew[r]);
        s[n][r] = p;
        lsum[r] += p;
      }
#pragma unroll
    for (int off = 1; off < 16; off <<= 1)
#pragma unroll
      for (int r = 0; r < 8; ++r)
        lsum[r] += __shfl_xor(lsum[r], off, 32);
#pragma unroll
    for (int r = 0; r < 8; ++r) l_i[r] = l_i[r] * rescale[r] + lsum[r];
#pragma unroll
    for (int n = 0; n < 4; ++n)
#pragma unroll
      for (int r = 0; r < 8; ++r) o[n][r] *= rescale[r];

    // ---- round-trip P through LDS to get it into A-fragment order --------
#pragma unroll
    for (int n = 0; n < 4; ++n)
#pragma unroll
      for (int r = 0; r < 8; ++r)
        Ps[w][(mg + r) * 64 + n * 16 + cn] = (_Float16)s[n][r];
    // wave-private LDS region; DS ops from one wave stay in order.

    const v16h ap0 = load_frag_a(&Ps[w][0], 64);    // keys j0+0..31
    const v16h ap1 = load_frag_a(&Ps[w][32], 64);   // keys j0+32..63

    // ---- O += P V --------------------------------------------------------
#pragma unroll
    for (int n = 0; n < 4; ++n) {
      v16h bv0 = load_frag_b(vbase + (long)j0 * HD + n * 16, HD);
      v16h bv1 = load_frag_b(vbase + (long)(j0 + 32) * HD + n * 16, HD);
      o[n] = WMMA_F16(ap0, bv0, o[n]);
      o[n] = WMMA_F16(ap1, bv1, o[n]);
    }
  }

  // ---- epilogue: O / l, store f16 in [B,S,NH*D] for the Wo GEMM ----------
  _Float16* obase = Oh + ((long)(b * S_LEN + qrow0)) * HID + h * HD;
#pragma unroll
  for (int n = 0; n < 4; ++n)
#pragma unroll
    for (int r = 0; r < 8; ++r) {
      const float val = o[n][r] / l_i[r];
      obase[(long)(mg + r) * HID + n * 16 + cn] = (_Float16)val;
    }
}

// ---------------------------------------------------------------------------
// Launch
// ---------------------------------------------------------------------------
extern "C" void kernel_launch(void* const* d_in, const int* in_sizes, int n_in,
                              void* d_out, int out_size, void* d_ws, size_t ws_size,
                              hipStream_t stream) {
  (void)in_sizes; (void)n_in; (void)out_size; (void)ws_size;

  const float* x  = (const float*)d_in[0];   // [8,512,2048]
  const float* Wq = (const float*)d_in[1];   // [2048,2048]
  const float* Wk = (const float*)d_in[2];   // [2048,512]
  const float* Wv = (const float*)d_in[3];   // [2048,512]
  const float* Wo = (const float*)d_in[4];   // [2048,2048]

  const long XN  = (long)B_SZ * S_LEN * HID;       // 8388608
  const long WQN = (long)HID * NH * HD;            // 4194304
  const long WKN = (long)HID * NKV * HD;           // 1048576
  const long KVN = (long)B_SZ * S_LEN * NKV * HD;  // 2097152

  float* out   = (float*)d_out;                    // [B,S,HID]
  float* k_out = out + XN;                         // [B,NKV,S,D]
  float* v_out = k_out + KVN;                      // [B,NKV,S,D]

  size_t off = 0;
  auto take = [&](long elems, size_t esize) -> void* {
    void* p = (char*)d_ws + off;
    off += ((size_t)elems * esize + 255) & ~(size_t)255;
    return p;
  };
  _Float16* x_h    = (_Float16*)take(XN,  2);
  _Float16* Wq_h   = (_Float16*)take(WQN, 2);
  _Float16* Wk_h   = (_Float16*)take(WKN, 2);
  _Float16* Wv_h   = (_Float16*)take(WKN, 2);
  _Float16* Wo_h   = (_Float16*)take(WQN, 2);
  float*    q_f32  = (float*)take(XN,  4);
  float*    k_f32  = (float*)take(KVN, 4);
  float*    v_f32  = (float*)take(KVN, 4);
  _Float16* q_h    = (_Float16*)take(XN,  2);
  _Float16* kt_h   = (_Float16*)take(KVN, 2);
  _Float16* v_h    = (_Float16*)take(KVN, 2);
  _Float16* attn_h = (_Float16*)take(XN,  2);

  // 1) f32 -> f16 casts
  cvt_f16_kernel<<<4096, 256, 0, stream>>>(x,  x_h,  XN);
  cvt_f16_kernel<<<4096, 256, 0, stream>>>(Wq, Wq_h, WQN);
  cvt_f16_kernel<<<2048, 256, 0, stream>>>(Wk, Wk_h, WKN);
  cvt_f16_kernel<<<2048, 256, 0, stream>>>(Wv, Wv_h, WKN);
  cvt_f16_kernel<<<4096, 256, 0, stream>>>(Wo, Wo_h, WQN);

  const int M = B_SZ * S_LEN;                      // 4096
  // 2) QKV projections
  gemm_f16_f32<<<dim3(HID / 128, M / 128), 256, 0, stream>>>(x_h, Wq_h, q_f32, M, HID, HID);
  gemm_f16_f32<<<dim3((NKV * HD) / 128, M / 128), 256, 0, stream>>>(x_h, Wk_h, k_f32, M, NKV * HD, HID);
  gemm_f16_f32<<<dim3((NKV * HD) / 128, M / 128), 256, 0, stream>>>(x_h, Wv_h, v_f32, M, NKV * HD, HID);

  // 3) RoPE + layout shuffles (+ f32 k/v cache outputs)
  rope_q_kernel<<<(int)(XN / 256), 256, 0, stream>>>(q_f32, q_h);
  rope_k_kernel<<<(int)(KVN / 256), 256, 0, stream>>>(k_f32, k_out, kt_h);
  copy_v_kernel<<<(int)(KVN / 256), 256, 0, stream>>>(v_f32, v_out, v_h);

  // 4) fused causal attention
  attn_kernel<<<dim3(S_LEN / 64, B_SZ * NH), 128, 0, stream>>>(q_h, kt_h, v_h, attn_h);

  // 5) output projection straight into d_out
  gemm_f16_f32<<<dim3(HID / 128, M / 128), 256, 0, stream>>>(attn_h, Wo_h, out, M, HID, HID);
}
